// MambaBlock_45483703664837
// MI455X (gfx1250) — matrix operand order repaired and checked by reading, not compiled, not run on hardware
//
#include <hip/hip_runtime.h>
#include <hip/hip_bf16.h>
#include <stdint.h>

// ---------------- problem constants (from reference) ----------------
#define B_   2
#define L_   2048
#define DM_  1024
#define DI_  2048
#define DR_  64
#define NS_  16        // state dim N
#define KC_  4         // conv kernel width
#define BL_  (B_ * L_) // 4096 rows for all GEMMs
#define NWX  96        // DR + 2N
#define NWXP 128       // padded rows for Wx (bf16)

// ---------------- WMMA vector types ----------------
typedef __bf16 v16bf __attribute__((ext_vector_type(16)));
typedef float  v8f   __attribute__((ext_vector_type(8)));

// f32 -> bf16 round-to-nearest-even
__device__ __forceinline__ __bf16 f2bf(float f) {
    union { float f; unsigned u; } v; v.f = f;
    unsigned r = v.u + 0x7FFFu + ((v.u >> 16) & 1u);
    unsigned short h = (unsigned short)(r >> 16);
    union { unsigned short s; __bf16 b; } o; o.s = h;
    return o.b;
}
__device__ __forceinline__ float bf2f(__bf16 b) {
    union { unsigned short s; __bf16 b; } i; i.b = b;
    union { unsigned u; float f; } o; o.u = ((unsigned)i.s) << 16;
    return o.f;
}
__device__ __forceinline__ float softplus_f(float x) {
    return (x > 20.f) ? x : log1pf(expf(x));
}
__device__ __forceinline__ float silu_f(float x) {
    return x / (1.f + expf(-x));
}

// ---------------- CDNA5 async global->LDS staging ----------------
#if defined(__AMDGCN__) && __has_builtin(__builtin_amdgcn_global_load_async_to_lds_b128) && \
    __has_builtin(__builtin_amdgcn_s_wait_asynccnt)
#define USE_ASYNC_LDS 1
#else
#define USE_ASYNC_LDS 0
#endif

// Builtin expects: (int4-vector __device__ (AS1) ptr, int4-vector __shared__ (AS3) ptr, Ii, Ii)
typedef int v4i __attribute__((vector_size(16)));
typedef __attribute__((address_space(1))) v4i* g_v4i_p;
typedef __attribute__((address_space(3))) v4i* l_v4i_p;

// Copy 16 bytes (8 bf16) global -> LDS. Async path bypasses VGPRs (ASYNCcnt).
// LDS AS3 cast via uintptr truncation is exact: flat LDS addr low 32 bits == LDS offset.
__device__ __forceinline__ void copy16_g2l(const __bf16* g, __bf16* l) {
#if USE_ASYNC_LDS
    __builtin_amdgcn_global_load_async_to_lds_b128(
        (g_v4i_p)(uintptr_t)g,
        (l_v4i_p)(uint32_t)(uintptr_t)l,
        0, 0);
#else
    *(uint4*)l = *(const uint4*)g;
#endif
}
// Wait until at most CNT async ops remain in flight (in-order completion per wave).
template <int CNT>
__device__ __forceinline__ void wait_g2l() {
#if USE_ASYNC_LDS
    __builtin_amdgcn_s_wait_asynccnt(CNT);
#endif
}

// ---------------- f32 -> bf16 convert (zero-pads [n_src, n_total)) ----------------
__global__ void __launch_bounds__(256)
cvt_f32_bf16(const float* __restrict__ src, __bf16* __restrict__ dst,
             int n_src, int n_total) {
    const int i = blockIdx.x * blockDim.x + threadIdx.x;
    if (i < n_total) dst[i] = (i < n_src) ? f2bf(src[i]) : f2bf(0.f);
}

// ---------------- WMMA GEMM: C[m,n] = act(sum_k A[m,k]*W[n,k] + bias[n]) ----------------
// A: bf16 (BL_ x lda), W: bf16 (Npad x K) with Npad >= gridDim.x*64.
// 64x64 tile / 256 threads (8 waves). K staged 32 at a time via DOUBLE-BUFFERED
// async global->LDS copies: stage k+1 is prefetched while stage k feeds the WMMA pipe;
// s_wait_asynccnt 2 retires only the older pair (<=4 in flight, in-order).
#define BM 64
#define BN 64
#define BK 32
#define LDP (BK + 8)   // 40 elems = 80B row stride: bank-spread, 16B-aligned fragments

__global__ void __launch_bounds__(256)
gemm_bf16_wmma(const __bf16* __restrict__ A, int lda,
               const __bf16* __restrict__ W,
               const float* __restrict__ bias,      // nullable
               float* __restrict__ C, int ldc,
               __bf16* __restrict__ Cb,             // nullable bf16 mirror (same ldc)
               int N, int Kdim, int act)            // act: 0=none, 1=softplus
{
    __shared__ __bf16 lds_a[2][BM][LDP];
    __shared__ __bf16 lds_b[2][BN][LDP];

    const int tid   = threadIdx.x;
    const int lane  = tid & 31;
    const int wave  = tid >> 5;
    const int mt    = wave & 3;    // m sub-tile 0..3
    const int npair = wave >> 2;   // n sub-tile pair 0..1
    const int half  = lane >> 4;
    const int l16   = lane & 15;

    const int mblock = blockIdx.y * BM;
    const int nblock = blockIdx.x * BN;

    // staging map: thread -> (row 0..63, 8-elem k-group 0..3); 16B per thread per tile
    const int trow = tid >> 2;
    const int tcol = (tid & 3) * 8;
    const __bf16* gA = A + (size_t)(mblock + trow) * lda  + tcol;
    const __bf16* gW = W + (size_t)(nblock + trow) * Kdim + tcol;
    __bf16* la[2] = { &lds_a[0][trow][tcol], &lds_a[1][trow][tcol] };
    __bf16* lb[2] = { &lds_b[0][trow][tcol], &lds_b[1][trow][tcol] };

    v8f acc0 = {};
    v8f acc1 = {};

    // prologue: prefetch stage 0
    copy16_g2l(gA, la[0]);
    copy16_g2l(gW, lb[0]);

    int cur = 0;
    for (int kk = 0; kk < Kdim; kk += BK) {
        const int nxt = kk + BK;
        if (nxt < Kdim) {
            // prefetch stage kk+BK into the other buffer, then retire stage kk's pair
            copy16_g2l(gA + nxt, la[cur ^ 1]);
            copy16_g2l(gW + nxt, lb[cur ^ 1]);
            wait_g2l<2>();
        } else {
            wait_g2l<0>();
        }
        __syncthreads();   // all waves' stage-kk copies have landed

        // A fragment (16x32): lane -> M=l16; K = half*16 + e
        const __bf16* pa = &lds_a[cur][mt * 16 + l16][half * 16];
        v16bf af;
        #pragma unroll
        for (int e = 0; e < 8; ++e) { af[e] = pa[e]; af[8 + e] = pa[8 + e]; }

        // B fragments (32x16): lane -> N=l16; K = half*16 + e
        const int n0 = (npair * 2) * 16 + l16;
        const __bf16* pb0 = &lds_b[cur][n0][half * 16];
        const __bf16* pb1 = &lds_b[cur][n0 + 16][half * 16];
        v16bf bf0, bf1;
        #pragma unroll
        for (int e = 0; e < 16; ++e) { bf0[e] = pb0[e]; bf1[e] = pb1[e]; }

        acc0 = __builtin_amdgcn_wmma_f32_16x16x32_bf16(
                   false, af, false, bf0, (short)0, acc0, false, false);
        acc1 = __builtin_amdgcn_wmma_f32_16x16x32_bf16(
                   false, af, false, bf1, (short)0, acc1, false, false);

        __syncthreads();   // end all reads of buf[cur] before it is re-used for stage kk+2*BK
        cur ^= 1;
    }

    // epilogue: VGPR r holds M = r + 8*half, N = l16
    const int ncol0 = nblock + (npair * 2) * 16 + l16;
    const int ncol1 = ncol0 + 16;
    const int mbase = mblock + mt * 16 + half * 8;
    #pragma unroll
    for (int r = 0; r < 8; ++r) {
        const int m = mbase + r;
        if (ncol0 < N) {
            float v = acc0[r];
            if (bias) v += bias[ncol0];
            if (act == 1) v = softplus_f(v);
            C[(size_t)m * ldc + ncol0] = v;
            if (Cb) Cb[(size_t)m * ldc + ncol0] = f2bf(v);
        }
        if (ncol1 < N) {
            float v = acc1[r];
            if (bias) v += bias[ncol1];
            if (act == 1) v = softplus_f(v);
            C[(size_t)m * ldc + ncol1] = v;
            if (Cb) Cb[(size_t)m * ldc + ncol1] = f2bf(v);
        }
    }
}

// ---------------- depthwise causal conv (K=4) + SiLU -> u (bf16) ----------------
__global__ void __launch_bounds__(256)
conv_silu_kernel(const float* __restrict__ xz,       // (B*L, 2*DI); xin = cols [0,DI)
                 const float* __restrict__ conv_w,   // (DI, K)
                 const float* __restrict__ conv_b,   // (DI,)
                 __bf16* __restrict__ u)             // (B*L, DI) bf16
{
    const size_t idx = (size_t)blockIdx.x * blockDim.x + threadIdx.x;
    const int d  = (int)(idx % DI_);
    const int bl = (int)(idx / DI_);
    const int l  = bl % L_;
    const int b  = bl / L_;

    float acc = conv_b[d];
    #pragma unroll
    for (int k = 0; k < KC_; ++k) {
        const int ls = l - (KC_ - 1) + k;
        const float xv = (ls >= 0)
            ? xz[((size_t)b * L_ + ls) * (2 * DI_) + d] : 0.f;
        acc = fmaf(xv, conv_w[d * KC_ + k], acc);
    }
    u[idx] = f2bf(silu_f(acc));
}

// ---------------- selective scan: one thread per (b,d), N=16 state in regs ----------------
__global__ void __launch_bounds__(256)
scan_kernel(const float* __restrict__ xz,     // z = cols [DI, 2*DI)
            const __bf16* __restrict__ u,     // (B*L, DI) bf16
            const float* __restrict__ dbc,    // (B*L, 96): dr|Bm|Cm (f32)
            const float* __restrict__ delta,  // (B*L, DI) f32
            const float* __restrict__ A_log,  // (DI, 16)
            const float* __restrict__ Dp,     // (DI,)
            __bf16* __restrict__ g)           // (B*L, DI) bf16 out
{
    const int tid = blockIdx.x * blockDim.x + threadIdx.x;
    if (tid >= B_ * DI_) return;
    const int d = tid % DI_;
    const int b = tid / DI_;

    float Arow[NS_];
    #pragma unroll
    for (int n = 0; n < NS_; ++n) Arow[n] = -expf(A_log[d * NS_ + n]);
    const float Dd = Dp[d];

    float h[NS_];
    #pragma unroll
    for (int n = 0; n < NS_; ++n) h[n] = 0.f;

    for (int l = 0; l < L_; ++l) {
        const size_t bl = (size_t)b * L_ + l;
        const float dl = delta[bl * DI_ + d];
        const float ul = bf2f(u[bl * DI_ + d]);
        const float du = dl * ul;
        const float* row = &dbc[bl * NWX];
        float y = 0.f;
        #pragma unroll
        for (int n = 0; n < NS_; ++n) {
            const float Bn = row[DR_ + n];
            const float Cn = row[DR_ + NS_ + n];
            h[n] = fmaf(expf(dl * Arow[n]), h[n], du * Bn);
            y = fmaf(h[n], Cn, y);
        }
        y = fmaf(Dd, ul, y);
        const float z = xz[bl * (2 * DI_) + DI_ + d];
        g[bl * DI_ + d] = f2bf(y * silu_f(z));
    }
}

// ---------------- host-side launcher ----------------
extern "C" void kernel_launch(void* const* d_in, const int* in_sizes, int n_in,
                              void* d_out, int out_size, void* d_ws, size_t ws_size,
                              hipStream_t stream) {
    const float* x      = (const float*)d_in[0];
    const float* Win    = (const float*)d_in[1];
    const float* conv_w = (const float*)d_in[2];
    const float* conv_b = (const float*)d_in[3];
    const float* Wx     = (const float*)d_in[4];
    const float* Wdt    = (const float*)d_in[5];
    const float* bdt    = (const float*)d_in[6];
    const float* A_log  = (const float*)d_in[7];
    const float* Dp     = (const float*)d_in[8];
    const float* Wout   = (const float*)d_in[9];
    float* out = (float*)d_out;

    // workspace carve-out (256B aligned chunks)
    char* p = (char*)d_ws;
    auto alloc = [&](size_t bytes) -> char* {
        char* r = p; p += (bytes + 255) & ~(size_t)255; return r;
    };
    float*  xz    = (float*)alloc((size_t)BL_ * 2 * DI_ * 4);   // 64 MB
    float*  delta = (float*)alloc((size_t)BL_ * DI_ * 4);       // 32 MB
    float*  dbc   = (float*)alloc((size_t)BL_ * NWX * 4);       // 1.5 MB
    __bf16* xb    = (__bf16*)alloc((size_t)BL_ * DM_ * 2);      // 8 MB
    __bf16* ub    = (__bf16*)alloc((size_t)BL_ * DI_ * 2);      // 16 MB
    __bf16* gb    = (__bf16*)alloc((size_t)BL_ * DI_ * 2);      // 16 MB
    __bf16* dbcb  = (__bf16*)alloc((size_t)BL_ * NWX * 2);      // 0.75 MB
    __bf16* WinB  = (__bf16*)alloc((size_t)2 * DI_ * DM_ * 2);  // 8 MB
    __bf16* WxB   = (__bf16*)alloc((size_t)NWXP * DI_ * 2);     // 0.5 MB (padded rows)
    __bf16* WdtB  = (__bf16*)alloc((size_t)DI_ * DR_ * 2);      // 0.25 MB
    __bf16* WoutB = (__bf16*)alloc((size_t)DM_ * DI_ * 2);      // 4 MB

    const dim3 blk(256);
    auto cvt = [&](const float* s, __bf16* d, int n_src, int n_total) {
        cvt_f32_bf16<<<(n_total + 255) / 256, blk, 0, stream>>>(s, d, n_src, n_total);
    };

    // 0) one-time-per-call bf16 conversions
    cvt(x,    xb,    BL_ * DM_,      BL_ * DM_);
    cvt(Win,  WinB,  2 * DI_ * DM_,  2 * DI_ * DM_);
    cvt(Wx,   WxB,   NWX * DI_,      NWXP * DI_);   // zero-pad rows 96..127
    cvt(Wdt,  WdtB,  DI_ * DR_,      DI_ * DR_);
    cvt(Wout, WoutB, DM_ * DI_,      DM_ * DI_);

    // 1) xz = x @ Win.T                 (N=4096, K=1024)
    gemm_bf16_wmma<<<dim3((2 * DI_) / BN, BL_ / BM), blk, 0, stream>>>(
        xb, DM_, WinB, nullptr, xz, 2 * DI_, nullptr, 2 * DI_, DM_, 0);

    // 2) u = silu(conv(xin))  -> bf16
    conv_silu_kernel<<<(B_ * L_ * DI_) / 256, blk, 0, stream>>>(xz, conv_w, conv_b, ub);

    // 3) dbc = u @ Wx.T                 (N=96 with padded W, K=2048), bf16 mirror for GEMM 4
    gemm_bf16_wmma<<<dim3(NWXP / BN, BL_ / BM), blk, 0, stream>>>(
        ub, DI_, WxB, nullptr, dbc, NWX, dbcb, NWX, DI_, 0);

    // 4) delta = softplus(dr @ Wdt.T + bdt)   (N=2048, K=64; lda=96)
    gemm_bf16_wmma<<<dim3(DI_ / BN, BL_ / BM), blk, 0, stream>>>(
        dbcb, NWX, WdtB, bdt, delta, DI_, nullptr, DI_, DR_, 1);

    // 5) selective scan -> g = y * silu(z)  (bf16)
    scan_kernel<<<(B_ * DI_ + 255) / 256, blk, 0, stream>>>(
        xz, ub, dbc, delta, A_log, Dp, gb);

    // 6) out = g @ Wout.T               (N=1024, K=2048)
    gemm_bf16_wmma<<<dim3(DM_ / BN, BL_ / BM), blk, 0, stream>>>(
        gb, DI_, WoutB, nullptr, out, DM_, nullptr, DM_, DI_, 0);
}
